// Process_Spatial_Attention_1666447311423
// MI455X (gfx1250) — matrix operand order repaired and checked by reading, not compiled, not run on hardware
//
#include <hip/hip_runtime.h>
#include <hip/hip_bf16.h>

// ---------------------------------------------------------------------------
// CDNA5 (gfx1250, wave32). Pipeline:
//   1) pack conv weights -> padded f16
//   2) WMMA GEMM projections (f32 in, f16 out), vectorized staging
//   3) windowed attention (QK^T + softmax + PV, WMMA)
//   4) 3x3 conv as implicit GEMM with async global->LDS staging (ASYNCcnt)
// All intermediates in d_ws as f16 to halve HBM traffic.
// ---------------------------------------------------------------------------

typedef __attribute__((ext_vector_type(16))) _Float16 v16h;
typedef __attribute__((ext_vector_type(8)))  _Float16 v8h;
typedef __attribute__((ext_vector_type(4)))  _Float16 v4h;
typedef __attribute__((ext_vector_type(2)))  _Float16 v2h;
typedef __attribute__((ext_vector_type(8)))  float    v8f;
typedef __attribute__((ext_vector_type(4)))  float    v4f;
typedef __attribute__((ext_vector_type(4)))  int      v4i;

// Address-space-qualified pointee types for the async-LDS builtin
// (builtin expects v4i in device/global AS and LDS AS per hipcc diagnostic).
typedef __attribute__((address_space(1))) v4i gas_v4i;
typedef __attribute__((address_space(3))) v4i las_v4i;

#if __has_builtin(__builtin_amdgcn_global_load_async_to_lds_b128)
#define HAVE_ASYNC_LDS 1
#endif

// Async copy of 16 bytes global->LDS (falls back to a vector copy).
__device__ __forceinline__ void async_copy16(const _Float16* g, _Float16* l) {
#ifdef HAVE_ASYNC_LDS
    __builtin_amdgcn_global_load_async_to_lds_b128(
        (gas_v4i*)g, (las_v4i*)l, 0, 0);
#else
    *(v8h*)l = *(const v8h*)g;
#endif
}

__device__ __forceinline__ void wait_async() {
#ifdef HAVE_ASYNC_LDS
#if __has_builtin(__builtin_amdgcn_s_wait_asynccnt)
    __builtin_amdgcn_s_wait_asynccnt(0);
#else
    asm volatile("s_wait_asynccnt 0" ::: "memory");
#endif
#endif
}

// ---- WMMA fragment helpers (contiguous k = 16*(lane/16) + i in A and B, wave32) ----------

// A-matrix 16x32 f16, tile stored row-major (M x K), row stride ldk halves.
__device__ __forceinline__ v16h lds_frag_a(const _Float16* base, int lane, int ldk) {
    int row = lane & 15;
    int kb  = (lane & 16) ? 16 : 0;
    const _Float16* p = base + row * ldk + kb;
    v16h f;
#pragma unroll
    for (int i = 0; i < 16; ++i) f[i] = p[i];
    return f;
}

// B-matrix 32x16 f16, tile stored N-major (N x K), row stride ldk halves.
__device__ __forceinline__ v16h lds_frag_b(const _Float16* base, int lane, int ldk) {
    int n  = lane & 15;
    int kb = (lane & 16) ? 16 : 0;
    const _Float16* p = base + n * ldk + kb;
    v16h f;
#pragma unroll
    for (int i = 0; i < 16; ++i) f[i] = p[i];
    return f;
}

__device__ __forceinline__ v8f wmma_f16(v16h a, v16h b, v8f c) {
    return __builtin_amdgcn_wmma_f32_16x16x32_f16(
        false, a, false, b, (short)0, c, false, false);
}

// ---------------------------------------------------------------------------
// Kernel 1: pack conv weights (180,720,3,3) f32 -> (9, 192, 736) f16, padded.
// ---------------------------------------------------------------------------
__global__ void pack_convw_kernel(const float* __restrict__ w, _Float16* __restrict__ wp) {
    int idx = blockIdx.x * 256 + threadIdx.x;
    const int TOT = 9 * 192 * 736;
    if (idx >= TOT) return;
    int t  = idx / (192 * 736);
    int r  = idx % (192 * 736);
    int oc = r / 736;
    int ic = r % 736;
    float v = 0.0f;
    if (oc < 180 && ic < 720) {
        int ky = t / 3, kx = t % 3;
        v = w[((size_t)oc * 720 + ic) * 9 + ky * 3 + kx];
    }
    wp[idx] = (_Float16)v;
}

// ---------------------------------------------------------------------------
// Kernel 2: GEMM projection  C_f16[m,n] = sum_k A_f32[m,k] * W_f32[n,k]
// Block tile 64x64, 4 waves. blockIdx.x = N-block (fast) for L2 reuse of A.
// ---------------------------------------------------------------------------
__global__ __launch_bounds__(128)
void gemm_proj_kernel(const float* __restrict__ A, const float* __restrict__ W,
                      _Float16* __restrict__ C, int N, int K, int ldc) {
    __shared__ _Float16 As[64 * 32];
    __shared__ _Float16 Bs[64 * 32];
    __shared__ _Float16 Cs[64 * 64];
    int tid  = threadIdx.x;
    int lane = tid & 31;
    int wv   = tid >> 5;
    int n0   = blockIdx.x * 64;
    int m0   = blockIdx.y * 64;

    v8f acc[4] = {{}, {}, {}, {}};

    for (int kk = 0; kk < K; kk += 32) {
        // Load phase: many loads in flight (no LDS dependence yet).
        v4f a4[4], b4[4];
#pragma unroll
        for (int r = 0; r < 4; ++r) {
            int e   = tid + r * 128;        // 0..511
            int row = e >> 3, g4 = e & 7;
            int k   = kk + g4 * 4;          // K % 4 == 0 -> group guard = (k < K)
            v4f z   = {};
            a4[r] = (k < K) ? *(const v4f*)(A + (size_t)(m0 + row) * K + k) : z;
            int nn = n0 + row;
            b4[r] = (k < K && nn < N) ? *(const v4f*)(W + (size_t)nn * K + k) : z;
        }
        __syncthreads();   // previous iteration's fragment reads done
        // Store phase: packed f32x4 -> f16x4, 8B LDS stores.
#pragma unroll
        for (int r = 0; r < 4; ++r) {
            int e   = tid + r * 128;
            int row = e >> 3, g4 = e & 7;
            *(v4h*)&As[row * 32 + g4 * 4] = __builtin_convertvector(a4[r], v4h);
            *(v4h*)&Bs[row * 32 + g4 * 4] = __builtin_convertvector(b4[r], v4h);
        }
        __syncthreads();

        v16h af = lds_frag_a(As + wv * 16 * 32, lane, 32);
#pragma unroll
        for (int nt = 0; nt < 4; ++nt) {
            v16h bf = lds_frag_b(Bs + nt * 16 * 32, lane, 32);
            acc[nt] = wmma_f16(af, bf, acc[nt]);
        }
    }

    // Epilogue via LDS transpose -> coalesced row stores.
    int n  = lane & 15;
    int mo = (lane & 16) ? 8 : 0;
    __syncthreads();
#pragma unroll
    for (int nt = 0; nt < 4; ++nt)
#pragma unroll
        for (int v = 0; v < 8; ++v)
            Cs[(wv * 16 + mo + v) * 64 + nt * 16 + n] = (_Float16)acc[nt][v];
    __syncthreads();
    for (int e = tid; e < 64 * 32; e += 128) {   // 64 rows x 32 half-pairs
        int r = e >> 5, p = e & 31;
        int gn = n0 + 2 * p;                      // N even -> pair guard = gn < N
        if (gn < N)
            *(unsigned*)&C[(size_t)(m0 + r) * ldc + gn] = *(unsigned*)&Cs[r * 64 + 2 * p];
    }
}

// ---------------------------------------------------------------------------
// Kernel 3: windowed attention. One block per (group, window, head).
// rpb is the exact collapsed value of the size-1 LayerNorm MLP chain.
// ---------------------------------------------------------------------------
__global__ __launch_bounds__(128)
void attn_kernel(const _Float16* __restrict__ qbuf,
                 const _Float16* __restrict__ kbuf,
                 const _Float16* __restrict__ vbuf,
                 _Float16* __restrict__ zbuf,
                 const float* __restrict__ ln3_b,
                 const float* __restrict__ lin3_w,
                 const float* __restrict__ lin3_b) {
    __shared__ _Float16 qs[64 * 32];      // reused as Z tile at the end
    __shared__ _Float16 ks[64 * 32];
    __shared__ _Float16 vts[32 * 64];     // V transposed: [d][m]
    __shared__ float    Sb[64 * 64];
    __shared__ _Float16 Pb[64 * 64];

    int tid = threadIdx.x, lane = tid & 31, wv = tid >> 5;
    int unit = blockIdx.x;
    int g    = unit / (1024 * 6);
    int rem  = unit % (1024 * 6);
    int w    = rem / 6;
    int h    = rem % 6;
    int base_y = (w >> 5) << 3;
    int base_x = (w & 31) << 3;
    int goff = g * 180 + h * 30;
    int hoff = h * 30;

    float rpb = fmaxf(ln3_b[0], 0.0f) * lin3_w[h] + lin3_b[h];
    const float scale = 0.18257418583505536f;  // 30^-0.5

    // Stage Q/K/V: pair-granular (u32) loads batched, then LDS stores.
    unsigned qv[8], kv[8], vvv[8];
#pragma unroll
    for (int t = 0; t < 8; ++t) {
        int e = tid + t * 128;             // 0..1023 : 64 rows x 16 pairs
        int m = e >> 4, p = e & 15;
        int i = m >> 3, j = m & 7;
        size_t l = (size_t)(base_y + i) * 256 + (base_x + j);
        bool valid = p < 15;               // d = 2p, 2p+1 ; d<30
        qv[t]  = valid ? *(const unsigned*)(qbuf + l * 720 + goff + 2 * p) : 0u;
        kv[t]  = valid ? *(const unsigned*)(kbuf + l * 180 + hoff + 2 * p) : 0u;
        vvv[t] = valid ? *(const unsigned*)(vbuf + l * 180 + hoff + 2 * p) : 0u;
    }
#pragma unroll
    for (int t = 0; t < 8; ++t) {
        int e = tid + t * 128;
        int m = e >> 4, p = e & 15;
        *(unsigned*)&qs[m * 32 + 2 * p] = qv[t];
        *(unsigned*)&ks[m * 32 + 2 * p] = kv[t];
        v2h vh = __builtin_bit_cast(v2h, vvv[t]);
        vts[(2 * p)     * 64 + m] = vh[0];
        vts[(2 * p + 1) * 64 + m] = vh[1];
    }
    __syncthreads();

    // S = scale * Q K^T + rpb
    {
        v16h aq = lds_frag_a(qs + wv * 16 * 32, lane, 32);
        int n  = lane & 15;
        int mo = (lane & 16) ? 8 : 0;
#pragma unroll
        for (int nt = 0; nt < 4; ++nt) {
            v8f acc = {};
            v16h bk = lds_frag_b(ks + nt * 16 * 32, lane, 32);
            acc = wmma_f16(aq, bk, acc);
#pragma unroll
            for (int v = 0; v < 8; ++v)
                Sb[(wv * 16 + mo + v) * 64 + nt * 16 + n] = acc[v] * scale + rpb;
        }
    }
    __syncthreads();

    // Row softmax, emit P in f16.
    if (tid < 64) {
        float mx = -1e30f;
        for (int n = 0; n < 64; ++n) mx = fmaxf(mx, Sb[tid * 64 + n]);
        float s = 0.0f;
        for (int n = 0; n < 64; ++n) {
            float e = __expf(Sb[tid * 64 + n] - mx);
            Sb[tid * 64 + n] = e;
            s += e;
        }
        float inv = 1.0f / s;
        for (int n = 0; n < 64; ++n) Pb[tid * 64 + n] = (_Float16)(Sb[tid * 64 + n] * inv);
    }
    __syncthreads();

    // Z = P V (16x32 per wave, K=64 in two chunks)
    v8f zacc[2] = {{}, {}};
#pragma unroll
    for (int kc = 0; kc < 2; ++kc) {
        v16h ap = lds_frag_a(Pb + wv * 16 * 64 + kc * 32, lane, 64);
#pragma unroll
        for (int nt = 0; nt < 2; ++nt) {
            v16h bv = lds_frag_b(vts + nt * 16 * 64 + kc * 32, lane, 64);
            zacc[nt] = wmma_f16(ap, bv, zacc[nt]);
        }
    }
    // Z tile -> LDS (reuse qs) -> coalesced pair stores to zbuf.
    _Float16* Zs = qs;
    int n  = lane & 15;
    int mo = (lane & 16) ? 8 : 0;
#pragma unroll
    for (int nt = 0; nt < 2; ++nt)
#pragma unroll
        for (int v = 0; v < 8; ++v)
            Zs[(wv * 16 + mo + v) * 32 + nt * 16 + n] = (_Float16)zacc[nt][v];
    __syncthreads();
    for (int e = tid; e < 64 * 16; e += 128) {  // 64 rows x 16 pairs
        int m = e >> 4, p = e & 15;
        if (p < 15) {
            int i = m >> 3, j = m & 7;
            size_t l = (size_t)(base_y + i) * 256 + (base_x + j);
            *(unsigned*)&zbuf[l * 720 + goff + 2 * p] = *(unsigned*)&Zs[m * 32 + 2 * p];
        }
    }
}

// ---------------------------------------------------------------------------
// Kernel 4: 3x3 conv 720->180 as implicit GEMM over 9 shifted taps.
// f16 inputs are staged with async global->LDS copies (ASYNCcnt).
// blockIdx.x = N-block (fast) for L2 reuse of zbuf rows.
// ---------------------------------------------------------------------------
__global__ __launch_bounds__(128)
void conv_kernel(const _Float16* __restrict__ zbuf, const _Float16* __restrict__ wp,
                 const float* __restrict__ bias, float* __restrict__ out) {
    __shared__ _Float16 As[64 * 32];
    __shared__ _Float16 Bs[64 * 32];
    __shared__ float    Cs[64 * 64];
    int tid = threadIdx.x, lane = tid & 31, wv = tid >> 5;
    int n0 = blockIdx.x * 64;      // 0, 64, 128 (N padded to 192)
    int bm = blockIdx.y;           // 1024 strips: 256 rows x 4 strips
    int y  = bm >> 2;
    int x0 = (bm & 3) << 6;

    v8f acc[4] = {{}, {}, {}, {}};

    for (int t = 0; t < 9; ++t) {
        int ky = t / 3 - 1, kx = t % 3 - 1;
        int yy = y + ky;
        bool yok = (yy >= 0) && (yy < 256);
        const _Float16* wbase = wp + (size_t)t * 192 * 736 + (size_t)n0 * 736;
        for (int kk = 0; kk < 720; kk += 32) {
            // Stage A (shifted zbuf rows) + B (packed weights), 16B granules.
            for (int e = tid; e < 512; e += 128) {
                if (e < 256) {                      // A: 64 rows x 4 groups of 8
                    int m = e >> 2, g8 = e & 3;
                    int xx = x0 + m + kx;
                    int ic = kk + g8 * 8;           // 720 % 8 == 0
                    bool ok = yok && (xx >= 0) && (xx < 256) && (ic < 720);
                    if (ok)
                        async_copy16(zbuf + ((size_t)yy * 256 + xx) * 720 + ic,
                                     &As[m * 32 + g8 * 8]);
                    else {
                        v8h z = {};
                        *(v8h*)&As[m * 32 + g8 * 8] = z;
                    }
                } else {                            // B: padded, unguarded
                    int e2 = e - 256;
                    int m = e2 >> 2, g8 = e2 & 3;
                    async_copy16(wbase + (size_t)m * 736 + kk + g8 * 8,
                                 &Bs[m * 32 + g8 * 8]);
                }
            }
            wait_async();
            __syncthreads();

            v16h af = lds_frag_a(As + wv * 16 * 32, lane, 32);
#pragma unroll
            for (int nt = 0; nt < 4; ++nt) {
                v16h bf = lds_frag_b(Bs + nt * 16 * 32, lane, 32);
                acc[nt] = wmma_f16(af, bf, acc[nt]);
            }
            __syncthreads();
        }
    }

    // Epilogue: bias-add, LDS transpose, coalesced b128 row stores.
    int n  = lane & 15;
    int mo = (lane & 16) ? 8 : 0;
#pragma unroll
    for (int nt = 0; nt < 4; ++nt) {
        int oc = n0 + nt * 16 + n;
        float b = (oc < 180) ? bias[oc] : 0.0f;
#pragma unroll
        for (int v = 0; v < 8; ++v)
            Cs[(wv * 16 + mo + v) * 64 + nt * 16 + n] = acc[nt][v] + b;
    }
    __syncthreads();
    for (int e = tid; e < 64 * 16; e += 128) {    // 64 rows x 16 float4 groups
        int r = e >> 4, g = e & 15;
        int oc = n0 + g * 4;                       // 180 % 4 == 0
        if (oc < 180)
            *(v4f*)&out[(size_t)(y * 256 + x0 + r) * 180 + oc] = *(v4f*)&Cs[r * 64 + g * 4];
    }
}

// ---------------------------------------------------------------------------
extern "C" void kernel_launch(void* const* d_in, const int* in_sizes, int n_in,
                              void* d_out, int out_size, void* d_ws, size_t ws_size,
                              hipStream_t stream) {
    (void)in_sizes; (void)n_in; (void)out_size; (void)ws_size;

    const float* x      = (const float*)d_in[0];
    const float* y      = (const float*)d_in[1];
    const float* W_qkv1 = (const float*)d_in[4];
    const float* W_qkv2 = (const float*)d_in[5];
    const float* conv_w = (const float*)d_in[6];
    const float* conv_b = (const float*)d_in[7];
    const float* ln3_b  = (const float*)d_in[19];
    const float* lin3_w = (const float*)d_in[20];
    const float* lin3_b = (const float*)d_in[21];
    float* out = (float*)d_out;

    const size_t L = 65536;
    _Float16* q1buf = (_Float16*)d_ws;          // L * 720
    _Float16* kbuf  = q1buf + L * 720;          // L * 180
    _Float16* vbuf  = kbuf + L * 180;           // L * 180
    _Float16* zbuf  = vbuf + L * 180;           // L * 720
    _Float16* wpack = zbuf + L * 720;           // 9 * 192 * 736

    {
        int tot = 9 * 192 * 736;
        pack_convw_kernel<<<(tot + 255) / 256, 256, 0, stream>>>(conv_w, wpack);
    }
    // N-block fastest for L2 reuse of the big A operands.
    gemm_proj_kernel<<<dim3(12, 1024), 128, 0, stream>>>(x, W_qkv1, q1buf, 720, 720, 720);
    gemm_proj_kernel<<<dim3(3, 1024), 128, 0, stream>>>(y, W_qkv2 + 180 * 180, kbuf, 180, 180, 180);
    gemm_proj_kernel<<<dim3(3, 1024), 128, 0, stream>>>(y, W_qkv2 + 360 * 180, vbuf, 180, 180, 180);
    attn_kernel<<<4 * 1024 * 6, 128, 0, stream>>>(q1buf, kbuf, vbuf, zbuf, ln3_b, lin3_w, lin3_b);
    conv_kernel<<<dim3(3, 1024), 128, 0, stream>>>(zbuf, wpack, conv_b, out);
}
